// GraphTransformerEncoder_80985903334154
// MI455X (gfx1250) — matrix operand order repaired and checked by reading, not compiled, not run on hardware
//
#include <hip/hip_runtime.h>
#include <hip/hip_bf16.h>

// ---------------------------------------------------------------------------
// Graph transformer encoder, CDNA5 (gfx1250, wave32).
// Dense projections via v_wmma_f32_16x16x32_bf16 with LDS layouts matched to
// the WMMA fragment layout (contiguous ds_load_b128 per fragment); edge
// attention via edge-parallel kernels with native f32 global atomics; fused
// beta-gate + GELU + residual + LayerNorm with wave32 shuffle reductions.
// ---------------------------------------------------------------------------

typedef __attribute__((ext_vector_type(16))) __bf16 v16bf;
typedef __attribute__((ext_vector_type(8)))  __bf16 v8bf;
typedef __attribute__((ext_vector_type(2)))  __bf16 v2bf;
typedef __attribute__((ext_vector_type(8)))  float  v8f;

#define NODES 50000
#define EDGES 800000
#define HIDD  128
#define NLAYER 3

struct GemmBatch {
    const float* W[4];
    const float* b[4];
    float*       out[4];
};

// ---------------------------------------------------------------------------
// out[sel] = A[M,K] @ W[sel][K,128] + b[sel]    (sel = blockIdx.y)
// block = 256 threads = 8 wave32; block computes 16 rows x 128 cols;
// wave w owns the 16x16 tile at columns [16w, 16w+16).
//
// LDS layouts are chosen so each lane's WMMA fragment is two contiguous
// 16-byte runs (ds_load_b128):
//   As[row][k]  row-major  (A frag: row=l15, K in {8h..8h+7} u {16+8h..16+8h+7})
//   Bs[col][k]  col-major  (B frag: col=l15+16w, K in {8h..8h+7} u {16+8h..16+8h+7})
// K is a template constant so the K-loop fully unrolls (4 or 8 WMMAs).
// ---------------------------------------------------------------------------
template <int K>
__global__ __launch_bounds__(256)
void gemm_bias_wmma_kernel(const float* __restrict__ A, GemmBatch gb)
{
    const int sel = blockIdx.y;
    const float* __restrict__ W    = gb.W[sel];
    const float* __restrict__ bias = gb.b[sel];
    float*       __restrict__ out  = gb.out[sel];

    __shared__ __align__(16) __bf16 As[16][40];    // 16 rows x 32 K (pad 40)
    __shared__ __align__(16) __bf16 Bs[128][40];   // 128 cols x 32 K (pad 40)

    const int  tid  = threadIdx.x;
    const int  lane = tid & 31;
    const int  wv   = tid >> 5;       // wave id 0..7 -> column tile
    const int  half = lane >> 4;      // 0 / 1
    const int  l15  = lane & 15;
    const long row0 = (long)blockIdx.x * 16;

    // staging assignments (fixed per thread)
    const int ar = tid >> 4;          // A: row 0..15
    const int ac = (tid & 15) * 2;    // A: col 0,2,..,30
    const int bcol  = tid & 127;      // B: column 0..127
    const int khalf = tid >> 7;       // B: k-half 0/1

    v8f acc = {};

    #pragma unroll
    for (int k0 = 0; k0 < K; k0 += 32) {
        // ---- stage A tile: 16x32 fp32 -> bf16 LDS (float2 load, packed store)
        {
            float2 f = *(const float2*)&A[(row0 + ar) * (long)K + k0 + ac];
            v2bf p;
            p[0] = (__bf16)f.x;
            p[1] = (__bf16)f.y;
            *(v2bf*)&As[ar][ac] = p;
        }
        // ---- stage W tile column-major: coalesced b32 loads, b128 LDS stores
        {
            float tmp[16];
            #pragma unroll
            for (int kk = 0; kk < 16; ++kk)
                tmp[kk] = W[(long)(k0 + khalf * 16 + kk) * HIDD + bcol];
            v8bf lo, hi;
            #pragma unroll
            for (int i = 0; i < 8; ++i) {
                lo[i] = (__bf16)tmp[i];
                hi[i] = (__bf16)tmp[8 + i];
            }
            *(v8bf*)&Bs[bcol][khalf * 16]     = lo;
            *(v8bf*)&Bs[bcol][khalf * 16 + 8] = hi;
        }
        if (k0 + 32 < K) {  // prefetch next chunk (global_prefetch_b8)
            __builtin_prefetch(&A[(row0 + ar) * (long)K + k0 + 32 + ac], 0, 0);
            __builtin_prefetch(&W[(long)(k0 + 32 + khalf) * HIDD + bcol], 0, 0);
        }
        __syncthreads();

        // ---- A fragment: two contiguous 16B runs -> ds_load_b128 x2
        v8bf a0 = *(const v8bf*)&As[l15][half * 8];
        v8bf a1 = *(const v8bf*)&As[l15][16 + half * 8];
        v16bf a = __builtin_shufflevector(a0, a1, 0, 1, 2, 3, 4, 5, 6, 7,
                                          8, 9, 10, 11, 12, 13, 14, 15);
        // ---- B fragment: one column, K in {8h..8h+7} u {16+8h..16+8h+7} -> ds_load_b128 x2
        v8bf b0 = *(const v8bf*)&Bs[wv * 16 + l15][half * 8];
        v8bf b1 = *(const v8bf*)&Bs[wv * 16 + l15][16 + half * 8];
        v16bf b = __builtin_shufflevector(b0, b1, 0, 1, 2, 3, 4, 5, 6, 7,
                                          8, 9, 10, 11, 12, 13, 14, 15);

        acc = __builtin_amdgcn_wmma_f32_16x16x32_bf16(
            false, a, false, b, (short)0, acc, false, false);
        __syncthreads();
    }

    // ---- epilogue: C/D layout -> row = r + 8*half, col = 16*wv + l15
    const int   col = wv * 16 + l15;
    const float bvv = bias[col];
    #pragma unroll
    for (int r = 0; r < 8; ++r) {
        long rr = row0 + r + 8 * half;
        out[rr * HIDD + col] = acc[r] + bvv;
    }
}

// ---------------------------------------------------------------------------
__global__ __launch_bounds__(256)
void fill_kernel(float* __restrict__ p, float v, int n)
{
    int i = blockIdx.x * 256 + threadIdx.x;
    if (i < n) p[i] = v;
}

__device__ inline void atomicMaxF(float* addr, float val)
{
    // ordered-bits trick; addr initialized to -FLT_MAX
    if (val >= 0.0f) atomicMax((int*)addr, __float_as_int(val));
    else             atomicMin((unsigned int*)addr, __float_as_uint(val));
}

// ---------------------------------------------------------------------------
// pass 1: alpha[e,h] = <q[dst], k[src]+e_emb> / sqrt(32);  segment max(dst)
// one thread per (edge, head); float4-vectorized dot product
// ---------------------------------------------------------------------------
__global__ __launch_bounds__(256)
void attn_logits_kernel(const float* __restrict__ q, const float* __restrict__ k,
                        const int* __restrict__ src, const int* __restrict__ dst,
                        const float* __restrict__ ea, const float* __restrict__ We_l,
                        float* __restrict__ exbuf, float* __restrict__ amax)
{
    int idx = blockIdx.x * 256 + threadIdx.x;          // < E*4
    int e = idx >> 2, h = idx & 3;
    int s = src[e], d = dst[e];
    float w = ea[e];
    const float4* qp = (const float4*)(q + (long)d * HIDD + h * 32);
    const float4* kp = (const float4*)(k + (long)s * HIDD + h * 32);
    const float4* wp = (const float4*)(We_l + h * 32);
    float acc = 0.0f;
    #pragma unroll
    for (int i = 0; i < 8; ++i) {
        float4 qv = qp[i], kv = kp[i], wv = wp[i];
        acc += qv.x * (kv.x + w * wv.x) + qv.y * (kv.y + w * wv.y) +
               qv.z * (kv.z + w * wv.z) + qv.w * (kv.w + w * wv.w);
    }
    float alpha = acc * 0.17677669529663689f;          // 1/sqrt(32)
    exbuf[idx] = alpha;
    atomicMaxF(&amax[d * 4 + h], alpha);
}

// ---------------------------------------------------------------------------
// pass 2: ex = exp(alpha - amax[dst]);  segment sum(dst)
// ---------------------------------------------------------------------------
__global__ __launch_bounds__(256)
void attn_exp_kernel(const int* __restrict__ dst,
                     const float* __restrict__ amax, float* __restrict__ denom,
                     float* __restrict__ exbuf)
{
    int idx = blockIdx.x * 256 + threadIdx.x;          // < E*4
    int e = idx >> 2, h = idx & 3;
    int d = dst[e];
    float ex = expf(exbuf[idx] - amax[d * 4 + h]);
    exbuf[idx] = ex;
    atomicAdd(&denom[d * 4 + h], ex);
}

// ---------------------------------------------------------------------------
// pass 3: out[dst, j] += attn * (v[src, j] + e_emb[j]);  thread per (edge, j)
// ---------------------------------------------------------------------------
__global__ __launch_bounds__(256)
void attn_scatter_kernel(const float* __restrict__ v,
                         const int* __restrict__ src, const int* __restrict__ dst,
                         const float* __restrict__ ea, const float* __restrict__ We_l,
                         const float* __restrict__ exbuf, const float* __restrict__ denom,
                         float* __restrict__ agg)
{
    long idx = (long)blockIdx.x * 256 + threadIdx.x;   // < E*128
    int e = (int)(idx >> 7), j = (int)(idx & 127), h = j >> 5;
    int s = src[e], d = dst[e];
    float attn = exbuf[e * 4 + h] / (denom[d * 4 + h] + 1e-16f);
    float val  = attn * (v[(long)s * HIDD + j] + ea[e] * We_l[j]);
    atomicAdd(&agg[(long)d * HIDD + j], val);
}

// ---------------------------------------------------------------------------
// fused epilogue: beta-gate + GELU(exact) + residual + LayerNorm.
// one wave32 per node, 4 features per lane (j = lane + 32*i).
// ---------------------------------------------------------------------------
__device__ inline float waveSum(float v)
{
    #pragma unroll
    for (int m = 16; m > 0; m >>= 1) v += __shfl_xor(v, m, 32);
    return v;
}

__global__ __launch_bounds__(256)
void node_epilogue_kernel(const float* __restrict__ agg, const float* __restrict__ skip,
                          const float* __restrict__ res, const float* __restrict__ Wb,
                          const float* __restrict__ lng, const float* __restrict__ lnb,
                          float* __restrict__ hout)
{
    int wv = threadIdx.x >> 5, lane = threadIdx.x & 31;
    int node = blockIdx.x * 8 + wv;                    // N % 8 == 0
    long base = (long)node * HIDD;

    float o[4], xr[4], r[4];
    float bsum = 0.0f;
    #pragma unroll
    for (int i = 0; i < 4; ++i) {
        int j = lane + 32 * i;
        o[i]  = agg[base + j];
        xr[i] = skip[base + j];
        r[i]  = res[base + j];
        bsum += o[i] * Wb[j] + xr[i] * Wb[HIDD + j] + (o[i] - xr[i]) * Wb[2 * HIDD + j];
    }
    bsum = waveSum(bsum);
    float beta = 1.0f / (1.0f + expf(-bsum));

    float t[4], s = 0.0f, s2 = 0.0f;
    #pragma unroll
    for (int i = 0; i < 4; ++i) {
        float hc = beta * xr[i] + (1.0f - beta) * o[i];
        float g  = 0.5f * hc * (1.0f + erff(hc * 0.70710678118654752f));  // exact GELU
        t[i] = g + r[i];
        s  += t[i];
        s2 += t[i] * t[i];
    }
    s  = waveSum(s);
    s2 = waveSum(s2);
    float mu   = s  * (1.0f / HIDD);
    float var  = s2 * (1.0f / HIDD) - mu * mu;
    float rstd = rsqrtf(var + 1e-5f);
    #pragma unroll
    for (int i = 0; i < 4; ++i) {
        int j = lane + 32 * i;
        hout[base + j] = (t[i] - mu) * rstd * lng[j] + lnb[j];
    }
}

// ---------------------------------------------------------------------------
extern "C" void kernel_launch(void* const* d_in, const int* in_sizes, int n_in,
                              void* d_out, int out_size, void* d_ws, size_t ws_size,
                              hipStream_t stream)
{
    const float* x     = (const float*)d_in[0];
    const int*   eidx  = (const int*)  d_in[1];
    const float* ea    = (const float*)d_in[2];
    const float* Wi    = (const float*)d_in[3];
    const float* bi    = (const float*)d_in[4];
    const float* Wq    = (const float*)d_in[5];
    const float* bq    = (const float*)d_in[6];
    const float* Wk    = (const float*)d_in[7];
    const float* bk    = (const float*)d_in[8];
    const float* Wv    = (const float*)d_in[9];
    const float* bv    = (const float*)d_in[10];
    const float* We    = (const float*)d_in[11];
    const float* Wsk   = (const float*)d_in[12];
    const float* bsk   = (const float*)d_in[13];
    const float* Wbeta = (const float*)d_in[14];
    const float* lng   = (const float*)d_in[15];
    const float* lnb   = (const float*)d_in[16];
    float*       out   = (float*)d_out;

    const int N = NODES, E = EDGES;
    const int* src = eidx;
    const int* dst = eidx + E;

    // workspace carving (floats)
    float* ws   = (float*)d_ws;
    float* hA   = ws;
    float* hB   = hA  + (size_t)N * HIDD;
    float* q    = hB  + (size_t)N * HIDD;
    float* k    = q   + (size_t)N * HIDD;
    float* v    = k   + (size_t)N * HIDD;
    float* sk   = v   + (size_t)N * HIDD;
    float* agg  = sk  + (size_t)N * HIDD;
    float* ex   = agg + (size_t)N * HIDD;
    float* amax = ex  + (size_t)E * 4;
    float* den  = amax + (size_t)N * 4;

    // ---- input projection: hA = x @ Wi + bi   (K = 256)
    {
        GemmBatch g{};
        g.W[0] = Wi; g.b[0] = bi; g.out[0] = hA;
        hipLaunchKernelGGL((gemm_bias_wmma_kernel<256>), dim3(N / 16, 1), dim3(256),
                           0, stream, x, g);
    }

    float* hin  = hA;
    float* hout = hB;
    for (int l = 0; l < NLAYER; ++l) {
        const float* Wq_l = Wq  + (size_t)l * HIDD * HIDD;
        const float* Wk_l = Wk  + (size_t)l * HIDD * HIDD;
        const float* Wv_l = Wv  + (size_t)l * HIDD * HIDD;
        const float* Ws_l = Wsk + (size_t)l * HIDD * HIDD;
        const float* We_l = We  + (size_t)l * HIDD;
        const float* Wb_l = Wbeta + (size_t)l * 3 * HIDD;

        // q/k/v/skip projections in one batched WMMA launch
        {
            GemmBatch g{};
            g.W[0] = Wq_l; g.b[0] = bq  + l * HIDD; g.out[0] = q;
            g.W[1] = Wk_l; g.b[1] = bk  + l * HIDD; g.out[1] = k;
            g.W[2] = Wv_l; g.b[2] = bv  + l * HIDD; g.out[2] = v;
            g.W[3] = Ws_l; g.b[3] = bsk + l * HIDD; g.out[3] = sk;
            hipLaunchKernelGGL((gemm_bias_wmma_kernel<128>), dim3(N / 16, 4), dim3(256),
                               0, stream, hin, g);
        }

        // segment-softmax state
        hipLaunchKernelGGL(fill_kernel, dim3((N * HIDD + 255) / 256), dim3(256), 0, stream,
                           agg, 0.0f, N * HIDD);
        hipLaunchKernelGGL(fill_kernel, dim3((N * 4 + 255) / 256), dim3(256), 0, stream,
                           den, 0.0f, N * 4);
        hipLaunchKernelGGL(fill_kernel, dim3((N * 4 + 255) / 256), dim3(256), 0, stream,
                           amax, -3.402823466e38f, N * 4);

        hipLaunchKernelGGL(attn_logits_kernel, dim3(E * 4 / 256), dim3(256), 0, stream,
                           q, k, src, dst, ea, We_l, ex, amax);
        hipLaunchKernelGGL(attn_exp_kernel, dim3(E * 4 / 256), dim3(256), 0, stream,
                           dst, amax, den, ex);
        hipLaunchKernelGGL(attn_scatter_kernel,
                           dim3((unsigned)((size_t)E * HIDD / 256)), dim3(256), 0, stream,
                           v, src, dst, ea, We_l, ex, den, agg);

        float* dest = (l == NLAYER - 1) ? out : hout;
        hipLaunchKernelGGL(node_epilogue_kernel, dim3(N / 8), dim3(256), 0, stream,
                           agg, sk, hin, Wb_l, lng + l * HIDD, lnb + l * HIDD, dest);

        float* tmp = hin; hin = dest; hout = tmp;
    }
    (void)in_sizes; (void)n_in; (void)out_size; (void)ws_size;
}
